// SelfAttention_68066641707418
// MI455X (gfx1250) — matrix operand order repaired and checked by reading, not compiled, not run on hardware
//
#include <hip/hip_runtime.h>
#include <hip/hip_bf16.h>

// ---------------------------------------------------------------------------
// CDNA5 (gfx1250) fused self-attention: bf16 WMMA + TDM (tensor_load_to_lds)
// ---------------------------------------------------------------------------

typedef __attribute__((ext_vector_type(16))) __bf16 v16bf;
typedef __attribute__((ext_vector_type(8)))  float  v8f;

typedef __attribute__((ext_vector_type(4))) unsigned int tdm_u32x4;
typedef __attribute__((ext_vector_type(8))) int          tdm_i32x8;
typedef __attribute__((ext_vector_type(4))) int          tdm_i32x4;

union Frag16 {           // 16 bf16 = 8 dwords (A 16x32 or B 32x16 fragment)
    v16bf v;
    unsigned int u[8];
};

#define NB_B      2
#define NB_T      2048
#define NB_C      1024
#define NB_H      16
#define NB_D      64
#define NB_M      (NB_B * NB_T)     // 4096 rows of X
#define NB_N1     (3 * NB_C)        // 3072 qkv cols

__device__ __forceinline__ unsigned short f2bf(float f) {
    unsigned int x = __float_as_uint(f);
    unsigned int r = (x + 0x7FFFu + ((x >> 16) & 1u)) >> 16;   // RNE
    return (unsigned short)r;
}

__device__ __forceinline__ v8f wmma_bf16(const Frag16& a, const Frag16& b, v8f c) {
    return __builtin_amdgcn_wmma_f32_16x16x32_bf16(
        false, a.v, false, b.v, (short)0, c, false, false);
}

__device__ __forceinline__ unsigned lds_off32(const void* p) {
    // LDS aperture: generic addr[31:0] is the LDS byte offset
    return (unsigned)(unsigned long long)p;
}

// 2D tile TDM load: global (bf16, row stride `stride0` elems) -> LDS (packed)
__device__ __forceinline__ void tdm_load_2d(unsigned lds_addr, const void* gptr,
                                            unsigned tile_d0, unsigned tile_d1,
                                            unsigned tensor_d0, unsigned tensor_d1,
                                            unsigned stride0) {
    unsigned long long ga = (unsigned long long)gptr;
    tdm_u32x4 g0;
    g0[0] = 1u;                                             // count=1, load desc
    g0[1] = lds_addr;                                       // lds_addr
    g0[2] = (unsigned)(ga & 0xFFFFFFFFull);                 // global_addr[31:0]
    g0[3] = (unsigned)((ga >> 32) & 0x1FFFFFFull)           // global_addr[56:32]
          | (2u << 30);                                     // type = 2 (image)
    tdm_i32x8 g1;
    g1[0] = (int)(1u << 16);                                // data_size=1 (2 bytes)
    g1[1] = (int)((tensor_d0 & 0xFFFFu) << 16);             // tensor_dim0[15:0]
    g1[2] = (int)(((tensor_d0 >> 16) & 0xFFFFu)
          | ((tensor_d1 & 0xFFFFu) << 16));                 // dim0[31:16] | dim1[15:0]
    g1[3] = (int)(((tensor_d1 >> 16) & 0xFFFFu)
          | ((tile_d0 & 0xFFFFu) << 16));                   // dim1[31:16] | tile_dim0
    g1[4] = (int)(tile_d1 & 0xFFFFu);                       // tile_dim1, tile_dim2=0
    g1[5] = (int)stride0;                                   // tensor_dim0_stride[31:0]
    g1[6] = 0;                                              // stride0[47:32], d1s[15:0]
    g1[7] = 0;
    tdm_i32x4 gz = {0, 0, 0, 0};
#if __clang_major__ >= 23
    tdm_i32x8 gz8 = {0, 0, 0, 0, 0, 0, 0, 0};
    __builtin_amdgcn_tensor_load_to_lds(g0, g1, gz, gz, gz8, 0);
#else
    __builtin_amdgcn_tensor_load_to_lds(g0, g1, gz, gz, 0);
#endif
}

// ---------------------------------------------------------------------------
// Converts
// ---------------------------------------------------------------------------
__global__ void cvt_f32_bf16(const float* __restrict__ in,
                             unsigned short* __restrict__ out, int n) {
    int i = blockIdx.x * blockDim.x + threadIdx.x;
    if (i < n) out[i] = f2bf(in[i]);
}

// Wt[n][k] = W[k][n]  (W is K x N row-major, K = 1024)
__global__ void cvt_transpose_bf16(const float* __restrict__ W,
                                   unsigned short* __restrict__ Wt,
                                   int N, int K) {
    int i = blockIdx.x * blockDim.x + threadIdx.x;
    if (i >= N * K) return;
    int n = i / K;
    int k = i - n * K;
    Wt[i] = f2bf(W[k * N + n]);
}

// ---------------------------------------------------------------------------
// WMMA GEMM mainloop: wave tile 32x64, block 256 thr (8 waves) => 256x64 tile.
// A-frag per-lane k = 16*half+2*v; B-frag k = 16*half+2*v.
// ---------------------------------------------------------------------------
#define GEMM_MAINLOOP(XPTR, WPTR, ACC)                                          \
    const unsigned short* xrow0 = (XPTR) + (size_t)(m0 + nl) * NB_C;            \
    const unsigned short* xrow1 = xrow0 + 16 * NB_C;                            \
    const unsigned short* wrow0 = (WPTR) + (size_t)(n0 + nl) * NB_C;            \
    for (int kk = 0; kk < NB_C; kk += 32) {                                     \
        Frag16 a0, a1, bf[4];                                                   \
        _Pragma("unroll")                                                       \
        for (int v = 0; v < 8; ++v) {                                           \
            int k = kk + (half << 4) + (v << 1);                                \
            a0.u[v] = *(const unsigned int*)(xrow0 + k);                        \
            a1.u[v] = *(const unsigned int*)(xrow1 + k);                        \
        }                                                                       \
        _Pragma("unroll")                                                       \
        for (int j = 0; j < 4; ++j) {                                           \
            const unsigned short* wr = wrow0 + (size_t)j * 16 * NB_C + kk;      \
            _Pragma("unroll")                                                   \
            for (int v = 0; v < 8; ++v)                                         \
                bf[j].u[v] = *(const unsigned int*)(wr + (half << 4) + (v << 1));\
        }                                                                       \
        __builtin_prefetch(xrow0 + kk + 32, 0, 3);                              \
        __builtin_prefetch(xrow1 + kk + 32, 0, 3);                              \
        __builtin_prefetch(wrow0 + kk + 32, 0, 3);                              \
        _Pragma("unroll")                                                       \
        for (int j = 0; j < 4; ++j) ACC[0][j] = wmma_bf16(a0, bf[j], ACC[0][j]);\
        _Pragma("unroll")                                                       \
        for (int j = 0; j < 4; ++j) ACC[1][j] = wmma_bf16(a1, bf[j], ACC[1][j]);\
    }

// GEMM1: Xbf[4096x1024] @ Wt_qkv(bf16 [3072][1024]) + b_qkv
//   -> Q (x0.125) [B,H,T,D], K [B,H,T,D], V -> Vt [B,H,D,T]   (all bf16)
__global__ __launch_bounds__(256) void gemm_qkv(
    const unsigned short* __restrict__ Xbf,
    const unsigned short* __restrict__ Wt,
    const float* __restrict__ bias,
    unsigned short* __restrict__ Qb,
    unsigned short* __restrict__ Kb,
    unsigned short* __restrict__ Vtb) {

    const int wave = threadIdx.x >> 5;
    const int lane = threadIdx.x & 31;
    const int nl   = lane & 15;
    const int half = lane >> 4;
    const int m0   = blockIdx.y * 256 + wave * 32;
    const int n0   = blockIdx.x * 64;

    v8f acc[2][4];
#pragma unroll
    for (int mi = 0; mi < 2; ++mi)
#pragma unroll
        for (int j = 0; j < 4; ++j)
#pragma unroll
            for (int i = 0; i < 8; ++i) acc[mi][j][i] = 0.0f;

    GEMM_MAINLOOP(Xbf, Wt, acc)

#pragma unroll
    for (int j = 0; j < 4; ++j) {
        int col   = n0 + j * 16 + nl;
        int which = col >> 10;               // 0=q 1=k 2=v
        int c     = col & 1023;
        int h     = c >> 6;
        int d     = c & 63;
        float bv  = bias[col];
#pragma unroll
        for (int mi = 0; mi < 2; ++mi)
#pragma unroll
            for (int v = 0; v < 8; ++v) {
                int r  = m0 + mi * 16 + v + (half << 3);
                int b_ = r >> 11;
                int t  = r & 2047;
                int bh = (b_ << 4) + h;
                float val = acc[mi][j][v] + bv;
                if (which == 0)
                    Qb[(((size_t)bh * NB_T + t) << 6) + d] = f2bf(val * 0.125f);
                else if (which == 1)
                    Kb[(((size_t)bh * NB_T + t) << 6) + d] = f2bf(val);
                else
                    Vtb[(((size_t)bh << 6) + d) * NB_T + t] = f2bf(val);
            }
    }
}

// ---------------------------------------------------------------------------
// Flash attention: TDM stages shared K/V tiles into LDS (double-buffered),
// 8 waves/block each own a 16-query tile of the same (b,h).
// grid = (B*H, T/128), block = 256
// ---------------------------------------------------------------------------
__global__ __launch_bounds__(256) void attn_kernel(
    const unsigned short* __restrict__ Qb,
    const unsigned short* __restrict__ Kb,
    const unsigned short* __restrict__ Vtb,
    unsigned short* __restrict__ Obf) {

    __shared__ unsigned short Klds[2][32][64];   // [buf][key][d]      4KB x2
    __shared__ unsigned short Vlds[2][64][32];   // [buf][d][key]      4KB x2
    __shared__ unsigned short Plds[8][16][36];   // per-wave P tile

    const int wave = threadIdx.x >> 5;
    const int lane = threadIdx.x & 31;
    const int nl   = lane & 15;
    const int half = lane >> 4;

    const int bh = blockIdx.x;                  // b*16 + h
    const int q0 = blockIdx.y * 128 + wave * 16;

    const unsigned short* Qp = Qb  + (size_t)bh * NB_T * NB_D;
    const unsigned short* Kp = Kb  + (size_t)bh * NB_T * NB_D;
    const unsigned short* Vp = Vtb + (size_t)bh * NB_D * NB_T;

    // Q A-fragments for d in [0,32) and [32,64)
    Frag16 qf[2];
    {
        const unsigned short* qrow = Qp + (size_t)(q0 + nl) * NB_D;
#pragma unroll
        for (int s = 0; s < 2; ++s)
#pragma unroll
            for (int v = 0; v < 8; ++v) {
                int d = s * 32 + (half << 4) + (v << 1);
                qf[s].u[v] = *(const unsigned int*)(qrow + d);
            }
    }

    // TDM preload: key blocks 0 and 32 into buffers 0 and 1 (wave 0 only)
    if (threadIdx.x < 32) {
#pragma unroll
        for (int pb = 0; pb < 2; ++pb) {
            int kbp = pb * 32;
            tdm_load_2d(lds_off32(&Klds[pb][0][0]),
                        Kp + (size_t)kbp * NB_D,
                        /*tile*/ NB_D, 32, /*tensor*/ NB_D, NB_T, NB_D);
            tdm_load_2d(lds_off32(&Vlds[pb][0][0]),
                        Vp + (size_t)kbp,
                        /*tile*/ 32, NB_D, /*tensor*/ NB_T, NB_D, NB_T);
        }
    }

    float mrow[8], lrow[8];
    v8f o[4];
#pragma unroll
    for (int v = 0; v < 8; ++v) { mrow[v] = -1e30f; lrow[v] = 0.0f; }
#pragma unroll
    for (int j = 0; j < 4; ++j)
#pragma unroll
        for (int i = 0; i < 8; ++i) o[j][i] = 0.0f;

    int buf = 0;
    for (int kb = 0; kb < NB_T; kb += 32) {
        if (threadIdx.x < 32) {
            if (kb + 64 < NB_T) __builtin_amdgcn_s_wait_tensorcnt(2);
            else                __builtin_amdgcn_s_wait_tensorcnt(0);
        }
        __syncthreads();

        // ---- scores S = Q(16x64) @ K^T(64x16) x2 key sub-tiles ----
        v8f s[2];
#pragma unroll
        for (int jn = 0; jn < 2; ++jn)
#pragma unroll
            for (int i = 0; i < 8; ++i) s[jn][i] = 0.0f;

#pragma unroll
        for (int jn = 0; jn < 2; ++jn) {
            const unsigned short* krow = &Klds[buf][jn * 16 + nl][0];
#pragma unroll
            for (int st = 0; st < 2; ++st) {
                Frag16 b;
#pragma unroll
                for (int v = 0; v < 8; ++v)
                    b.u[v] = *(const unsigned int*)(krow + st * 32 + (half << 4) + (v << 1));
                s[jn] = wmma_bf16(qf[st], b, s[jn]);
            }
        }

        // ---- online softmax (row m = v + 8*half; 16 cols across lanes) ----
        float p0[8], p1[8];
#pragma unroll
        for (int v = 0; v < 8; ++v) {
            float mx = fmaxf(s[0][v], s[1][v]);
#pragma unroll
            for (int off = 8; off >= 1; off >>= 1)
                mx = fmaxf(mx, __shfl_xor(mx, off, 16));
            float mnew = fmaxf(mrow[v], mx);
            float corr = __expf(mrow[v] - mnew);
            float e0 = __expf(s[0][v] - mnew);
            float e1 = __expf(s[1][v] - mnew);
            float sum = e0 + e1;
#pragma unroll
            for (int off = 8; off >= 1; off >>= 1)
                sum += __shfl_xor(sum, off, 16);
            lrow[v] = lrow[v] * corr + sum;
            mrow[v] = mnew;
            p0[v] = e0;
            p1[v] = e1;
#pragma unroll
            for (int j = 0; j < 4; ++j) o[j][v] *= corr;
        }

        // ---- transpose P (C-layout -> A-layout) through per-wave LDS ----
#pragma unroll
        for (int v = 0; v < 8; ++v) {
            int r = v + (half << 3);
            Plds[wave][r][nl]      = f2bf(p0[v]);
            Plds[wave][r][16 + nl] = f2bf(p1[v]);
        }
        Frag16 pa;
#pragma unroll
        for (int v = 0; v < 8; ++v) {
            int k = (half << 4) + (v << 1);
            pa.u[v] = *(const unsigned int*)(&Plds[wave][nl][k]);
        }

        // ---- O += P(16x32) @ V(32x64) ----
#pragma unroll
        for (int jd = 0; jd < 4; ++jd) {
            Frag16 vb;
            const unsigned short* vrow = &Vlds[buf][jd * 16 + nl][half << 4];
#pragma unroll
            for (int v = 0; v < 8; ++v)
                vb.u[v] = *(const unsigned int*)(vrow + (v << 1));
            o[jd] = wmma_bf16(pa, vb, o[jd]);
        }

        __syncthreads();   // all waves done with buf before TDM overwrites it

        if (threadIdx.x < 32 && (kb + 64) < NB_T) {
            int kbn = kb + 64;
            tdm_load_2d(lds_off32(&Klds[buf][0][0]),
                        Kp + (size_t)kbn * NB_D,
                        NB_D, 32, NB_D, NB_T, NB_D);
            tdm_load_2d(lds_off32(&Vlds[buf][0][0]),
                        Vp + (size_t)kbn,
                        32, NB_D, NB_T, NB_D, NB_T);
        }
        buf ^= 1;
    }

    // ---- epilogue: O /= l, write bf16 [B,T,C] ----
    const int b_ = bh >> 4;
    const int h  = bh & 15;
#pragma unroll
    for (int v = 0; v < 8; ++v) {
        int t = q0 + v + (half << 3);
        float inv = 1.0f / lrow[v];
        size_t rowoff = ((size_t)(b_ * NB_T + t)) * NB_C + h * NB_D;
#pragma unroll
        for (int jd = 0; jd < 4; ++jd)
            Obf[rowoff + jd * 16 + nl] = f2bf(o[jd][v] * inv);
    }
}

// ---------------------------------------------------------------------------
// GEMM2: Obf[4096x1024] @ Wt_out(bf16 [1024][1024]) + b_out -> f32 out
// ---------------------------------------------------------------------------
__global__ __launch_bounds__(256) void gemm_out(
    const unsigned short* __restrict__ Obf,
    const unsigned short* __restrict__ Wt,
    const float* __restrict__ bias,
    float* __restrict__ out) {

    const int wave = threadIdx.x >> 5;
    const int lane = threadIdx.x & 31;
    const int nl   = lane & 15;
    const int half = lane >> 4;
    const int m0   = blockIdx.y * 256 + wave * 32;
    const int n0   = blockIdx.x * 64;

    v8f acc[2][4];
#pragma unroll
    for (int mi = 0; mi < 2; ++mi)
#pragma unroll
        for (int j = 0; j < 4; ++j)
#pragma unroll
            for (int i = 0; i < 8; ++i) acc[mi][j][i] = 0.0f;

    GEMM_MAINLOOP(Obf, Wt, acc)

#pragma unroll
    for (int j = 0; j < 4; ++j) {
        int col  = n0 + j * 16 + nl;
        float bv = bias[col];
#pragma unroll
        for (int mi = 0; mi < 2; ++mi)
#pragma unroll
            for (int v = 0; v < 8; ++v) {
                int r = m0 + mi * 16 + v + (half << 3);
                out[(size_t)r * NB_C + col] = acc[mi][j][v] + bv;
            }
    }
}

// ---------------------------------------------------------------------------
// Launch
// ---------------------------------------------------------------------------
extern "C" void kernel_launch(void* const* d_in, const int* in_sizes, int n_in,
                              void* d_out, int out_size, void* d_ws, size_t ws_size,
                              hipStream_t stream) {
    const float* x     = (const float*)d_in[0];   // [2,2048,1024]
    const float* W_qkv = (const float*)d_in[1];   // [1024,3072]
    const float* b_qkv = (const float*)d_in[2];   // [3072]
    const float* W_out = (const float*)d_in[3];   // [1024,1024]
    const float* b_out = (const float*)d_in[4];   // [1024]
    float* out = (float*)d_out;                   // [2,2048,1024] f32

    char* ws = (char*)d_ws;
    const size_t szXbf   = (size_t)NB_M * NB_C * 2;
    const size_t szWtqkv = (size_t)NB_N1 * NB_C * 2;
    const size_t szWtout = (size_t)NB_C * NB_C * 2;
    const size_t szQ     = (size_t)NB_B * NB_H * NB_T * NB_D * 2;

    unsigned short* Xbf   = (unsigned short*)(ws);
    unsigned short* Obf   = Xbf;                                   // alias
    unsigned short* Wtqkv = (unsigned short*)(ws + szXbf);
    unsigned short* Wtout = (unsigned short*)(ws + szXbf + szWtqkv);
    unsigned short* Qb    = (unsigned short*)(ws + szXbf + szWtqkv + szWtout);
    unsigned short* Kb    = (unsigned short*)((char*)Qb + szQ);
    unsigned short* Vtb   = (unsigned short*)((char*)Kb + szQ);

    {
        int n = NB_M * NB_C;
        cvt_f32_bf16<<<(n + 255) / 256, 256, 0, stream>>>(x, Xbf, n);
    }
    {
        int n = NB_N1 * NB_C;
        cvt_transpose_bf16<<<(n + 255) / 256, 256, 0, stream>>>(W_qkv, Wtqkv, NB_N1, NB_C);
    }
    {
        int n = NB_C * NB_C;
        cvt_transpose_bf16<<<(n + 255) / 256, 256, 0, stream>>>(W_out, Wtout, NB_C, NB_C);
    }

    gemm_qkv<<<dim3(NB_N1 / 64, NB_M / 256), 256, 0, stream>>>(
        Xbf, Wtqkv, b_qkv, Qb, Kb, Vtb);

    attn_kernel<<<dim3(NB_B * NB_H, NB_T / 128), 256, 0, stream>>>(
        Qb, Kb, Vtb, Obf);

    gemm_out<<<dim3(NB_C / 64, NB_M / 256), 256, 0, stream>>>(
        Obf, Wtout, b_out, out);
}
